// TransformerBlock_53060025975284
// MI455X (gfx1250) — matrix operand order repaired and checked by reading, not compiled, not run on hardware
//
#include <hip/hip_runtime.h>
#include <hip/hip_bf16.h>
#include <math.h>

// ---------------------------------------------------------------------------
// Sizes fixed by the reference: B=4, C=64, QC=32, H=W=64, L=4096, D=9*QC=288
// ---------------------------------------------------------------------------
#define BB 4
#define CC 64
#define QC 32
#define HH 64
#define WW 64
#define LL (HH * WW)
#define DD (9 * QC)     // 288
#define KCH (DD / 32)   // 9 chunks of K=32 per WMMA
#define NSEG 4          // key-dimension split for occupancy
#define SEGL (LL / NSEG)            // 1024 keys per segment
#define SEGT (SEGL / 16)            // 64 key tiles per segment
#define TILE_BYTES (16 * DD * 2)    // 9216 contiguous bytes per 16-row K tile
#define TILE_CHUNKS (TILE_BYTES / 16)  // 576 x b128
#define LDS_ROW_H 296               // padded LDS row stride in halves (592 B)

typedef __attribute__((ext_vector_type(16))) _Float16 v16h;
typedef __attribute__((ext_vector_type(8)))  _Float16 v8h;
typedef __attribute__((ext_vector_type(8)))  float    v8f;

// ---------------------------------------------------------------------------
// Generic 3x3 same-conv, NCHW/OIHW, optional pre-conv residual add + ReLU.
// ---------------------------------------------------------------------------
__global__ __launch_bounds__(256) void conv3_kernel(
    const float* __restrict__ in, const float* __restrict__ addin,
    const float* __restrict__ w,  const float* __restrict__ bias,
    float* __restrict__ out, int Cin, int Cout, int doRelu)
{
  int idx = blockIdx.x * blockDim.x + threadIdx.x;      // B*Cout*H*W
  int x  = idx & (WW - 1);
  int y  = (idx >> 6) & (HH - 1);
  int co = (idx >> 12) % Cout;
  int b  = idx / (Cout * LL);

  float acc = bias[co];
  for (int ci = 0; ci < Cin; ++ci) {
    const float* ip = in + ((size_t)(b * Cin + ci)) * LL;
    const float* ap = addin ? addin + ((size_t)(b * Cin + ci)) * LL : nullptr;
    const float* wp = w + (size_t)(co * Cin + ci) * 9;
#pragma unroll
    for (int i = 0; i < 3; ++i) {
#pragma unroll
      for (int j = 0; j < 3; ++j) {
        int yy = y + i - 1, xx = x + j - 1;
        if (yy >= 0 && yy < HH && xx >= 0 && xx < WW) {
          float v = ip[yy * WW + xx];
          if (ap) v += ap[yy * WW + xx];
          acc = fmaf(v, wp[i * 3 + j], acc);
        }
      }
    }
  }
  if (doRelu) acc = fmaxf(acc, 0.0f);
  out[idx] = acc;
}

// ---------------------------------------------------------------------------
// unfold3 + L2-normalize + f32->f16, row-major [B][L][288], d = c*9+i*3+j.
// ---------------------------------------------------------------------------
__global__ __launch_bounds__(256) void unfold_norm_f16_kernel(
    const float* __restrict__ src, _Float16* __restrict__ dst)
{
  int t = blockIdx.x * blockDim.x + threadIdx.x;        // B*L
  int l = t & (LL - 1);
  int b = t >> 12;
  int y = l >> 6, x = l & (WW - 1);
  const float* sb = src + (size_t)b * QC * LL;

  float nrm = 0.0f;
  for (int c = 0; c < QC; ++c) {
    const float* sc = sb + (size_t)c * LL;
#pragma unroll
    for (int i = 0; i < 3; ++i)
#pragma unroll
      for (int j = 0; j < 3; ++j) {
        int yy = y + i - 1, xx = x + j - 1;
        float v = (yy >= 0 && yy < HH && xx >= 0 && xx < WW) ? sc[yy * WW + xx] : 0.0f;
        nrm = fmaf(v, v, nrm);
      }
  }
  float inv = 1.0f / fmaxf(sqrtf(nrm), 1e-12f);

  _Float16* op = dst + (size_t)t * DD;
  for (int c = 0; c < QC; ++c) {
    const float* sc = sb + (size_t)c * LL;
#pragma unroll
    for (int i = 0; i < 3; ++i)
#pragma unroll
      for (int j = 0; j < 3; ++j) {
        int yy = y + i - 1, xx = x + j - 1;
        float v = (yy >= 0 && yy < HH && xx >= 0 && xx < WW) ? sc[yy * WW + xx] : 0.0f;
        op[c * 9 + i * 3 + j] = (_Float16)(v * inv);
      }
  }
}

// ---------------------------------------------------------------------------
// Async global->LDS staging of one 16-row key tile (9216 contiguous bytes),
// spread across 256 threads as b128 chunks (2-3 per thread). ASYNCcnt-tracked.
// ---------------------------------------------------------------------------
__device__ __forceinline__ void issue_tile_async(
    const _Float16* kseg_base, int lt, unsigned lds_base)
{
  const int tid = threadIdx.x;
  const unsigned g0 = (unsigned)(lt * TILE_BYTES);
  const unsigned long long sb = (unsigned long long)(uintptr_t)kseg_base;
#pragma unroll
  for (int i = 0; i < 3; ++i) {
    int c = tid + i * 256;
    if (c < TILE_CHUNKS) {                 // i==2 only active for tid<64
      unsigned row  = (unsigned)c / 36u;
      unsigned col  = (unsigned)c % 36u;
      unsigned loff = lds_base + row * (LDS_ROW_H * 2) + col * 16u;
      unsigned goff = g0 + (unsigned)c * 16u;
      asm volatile("global_load_async_to_lds_b128 %0, %1, %2 offset:0"
                   :: "v"(loff), "v"(goff), "s"(sb)
                   : "memory");
    }
  }
}

// ---------------------------------------------------------------------------
// Fused affinity GEMM + running max/argmax over a key segment.
// 256 threads = 8 waves; each wave owns 32 queries (two resident B-fragment
// sets, 18 x v16h). Key tiles staged via async-to-LDS, double buffered; each
// A-fragment (from ds_load_b128) feeds two V_WMMA_F32_16X16X32_F16.
// Grid: B * (L/256 query groups) * NSEG key segments = 256 blocks.
// ---------------------------------------------------------------------------
__global__ __launch_bounds__(256) void attn_argmax_kernel(
    const _Float16* __restrict__ Qmat,  // [B][L][288] normalized queries
    const _Float16* __restrict__ Kmat,  // [B][L][288] normalized keys
    float* __restrict__ pVal, int* __restrict__ pIdx)   // [B][NSEG][L]
{
  __shared__ __align__(16) _Float16 lds[2][16][LDS_ROW_H];

  const int wave = threadIdx.x >> 5;
  const int lane = threadIdx.x & 31;
  const int hi   = lane >> 4;
  const int lr   = lane & 15;

  const int bid = blockIdx.x;                 // ((b*16)+qg)*4 + seg
  const int seg = bid & (NSEG - 1);
  const int qg  = (bid >> 2) & 15;
  const int b   = bid >> 6;

  const int qbase = qg * 256 + wave * 32;     // 32 queries per wave
  const _Float16* Qb   = Qmat + (size_t)b * LL * DD;
  const _Float16* Kseg = Kmat + (size_t)b * LL * DD + (size_t)seg * SEGL * DD;

  // --- resident B fragments for two query tiles ---------------------------
  // B k order used here (contiguous): lanes 0-15 hold K=0..15, lanes 16-31 hold K=16..31.
  v16h bf0[KCH], bf1[KCH];
  {
    const _Float16* q0 = Qb + (size_t)(qbase + lr) * DD;
    const _Float16* q1 = Qb + (size_t)(qbase + 16 + lr) * DD;
#pragma unroll
    for (int kk = 0; kk < KCH; ++kk) {
      const v8h* p0 = (const v8h*)(q0 + kk * 32 + hi * 16);
      const v8h* p1 = (const v8h*)(q1 + kk * 32 + hi * 16);
      bf0[kk] = __builtin_shufflevector(p0[0], p0[1], 0, 1, 2, 3, 4, 5, 6, 7,
                                                      8, 9, 10, 11, 12, 13, 14, 15);
      bf1[kk] = __builtin_shufflevector(p1[0], p1[1], 0, 1, 2, 3, 4, 5, 6, 7,
                                                      8, 9, 10, 11, 12, 13, 14, 15);
    }
  }

  float bv0 = -2.0f, bv1 = -2.0f;             // unit dots are in [-1, 1]
  int   bi0 = seg * SEGL, bi1 = seg * SEGL;

  const unsigned ldsAddr[2] = {
    (unsigned)(uintptr_t)&lds[0][0][0],
    (unsigned)(uintptr_t)&lds[1][0][0],
  };

  issue_tile_async(Kseg, 0, ldsAddr[0]);

  for (int lt = 0; lt < SEGT; ++lt) {
    // my async loads for tile lt have landed in LDS
    asm volatile("s_wait_asynccnt 0x0" ::: "memory");
    // everyone's loads landed; everyone done reading the other buffer
    __syncthreads();
    if (lt + 1 < SEGT) issue_tile_async(Kseg, lt + 1, ldsAddr[(lt + 1) & 1]);

    const _Float16* lrow = &lds[lt & 1][lr][0];
    v8f acc0 = {}, acc1 = {};
#pragma unroll
    for (int kk = 0; kk < KCH; ++kk) {
      // A k order used here (contiguous, as B): half 0: K {0..15}; half 1: K {16..31}
      const v8h* a0 = (const v8h*)(lrow + kk * 32 + 16 * hi);
      const v8h* a1 = (const v8h*)(lrow + kk * 32 + 16 * hi + 8);
      v16h af = __builtin_shufflevector(a0[0], a1[0], 0, 1, 2, 3, 4, 5, 6, 7,
                                                      8, 9, 10, 11, 12, 13, 14, 15);
      acc0 = __builtin_amdgcn_wmma_f32_16x16x32_f16(
                 false, af, false, bf0[kk], (short)0, acc0, false, false);
      acc1 = __builtin_amdgcn_wmma_f32_16x16x32_f16(
                 false, af, false, bf1[kk], (short)0, acc1, false, false);
    }
    // D layout: lane<16 -> M=r, lane>=16 -> M=r+8 ; N = lane&15 (query col).
#pragma unroll
    for (int r = 0; r < 8; ++r) {
      int l = seg * SEGL + lt * 16 + hi * 8 + r;   // ascending scan, strict >
      float v0 = acc0[r];
      if (v0 > bv0) { bv0 = v0; bi0 = l; }
      float v1 = acc1[r];
      if (v1 > bv1) { bv1 = v1; bi1 = l; }
    }
  }

  // Merge the two lane halves covering the same query; lower index wins ties.
  {
    float o = __shfl_xor(bv0, 16, 32);
    int   j = __shfl_xor(bi0, 16, 32);
    if (o > bv0 || (o == bv0 && j < bi0)) { bv0 = o; bi0 = j; }
    o = __shfl_xor(bv1, 16, 32);
    j = __shfl_xor(bi1, 16, 32);
    if (o > bv1 || (o == bv1 && j < bi1)) { bv1 = o; bi1 = j; }
  }
  if (hi == 0) {
    size_t base = ((size_t)(b * NSEG + seg)) * LL;
    pVal[base + qbase + lr]      = bv0;
    pIdx[base + qbase + lr]      = bi0;
    pVal[base + qbase + 16 + lr] = bv1;
    pIdx[base + qbase + 16 + lr] = bi1;
  }
}

// ---------------------------------------------------------------------------
// Merge key-segment partials (ascending seg => first-argmax preserved).
// ---------------------------------------------------------------------------
__global__ __launch_bounds__(256) void reduce_seg_kernel(
    const float* __restrict__ pVal, const int* __restrict__ pIdx,
    float* __restrict__ Rmax, int* __restrict__ Ridx)
{
  int t = blockIdx.x * blockDim.x + threadIdx.x;   // B*L
  int b = t >> 12;
  int m = t & (LL - 1);
  float bv = -2.0f;
  int   bi = 0;
#pragma unroll
  for (int s = 0; s < NSEG; ++s) {
    size_t base = ((size_t)(b * NSEG + s)) * LL + m;
    float v = pVal[base];
    int   i = pIdx[base];
    if (v > bv || (v == bv && i < bi)) { bv = v; bi = i; }
  }
  Rmax[t] = bv;
  Ridx[t] = bi;
}

// ---------------------------------------------------------------------------
// Epilogue: T = fold3(gather(unfold3(v), R_idx))/9 ; out = f2 + x + T*S.
// ---------------------------------------------------------------------------
__global__ __launch_bounds__(256) void epilogue_kernel(
    const float* __restrict__ x, const float* __restrict__ v,
    const float* __restrict__ f2, const float* __restrict__ Rmax,
    const int* __restrict__ Ridx, float* __restrict__ out)
{
  int idx = blockIdx.x * blockDim.x + threadIdx.x;     // B*C*H*W
  int xq = idx & (WW - 1);
  int y  = (idx >> 6) & (HH - 1);
  int c  = (idx >> 12) & (CC - 1);
  int b  = idx >> 18;

  const float* vb  = v + ((size_t)(b * CC + c)) * LL;
  const int*   rib = Ridx + b * LL;

  float t = 0.0f;
#pragma unroll
  for (int i = 0; i < 3; ++i) {
#pragma unroll
    for (int j = 0; j < 3; ++j) {
      int py = y + 1 - i, px = xq + 1 - j;
      if (py < 0 || py >= HH || px < 0 || px >= WW) continue;
      int lp = rib[py * WW + px];
      int sy = (lp >> 6) + i - 1;
      int sx = (lp & (WW - 1)) + j - 1;
      if (sy < 0 || sy >= HH || sx < 0 || sx >= WW) continue;
      t += vb[sy * WW + sx];
    }
  }
  t *= (1.0f / 9.0f);
  float S = Rmax[b * LL + y * WW + xq];
  out[idx] = f2[idx] + x[idx] + t * S;
}

// ---------------------------------------------------------------------------
// Host-side orchestration (graph-capture safe: only kernel launches on stream)
// ---------------------------------------------------------------------------
extern "C" void kernel_launch(void* const* d_in, const int* in_sizes, int n_in,
                              void* d_out, int out_size, void* d_ws, size_t ws_size,
                              hipStream_t stream) {
  (void)in_sizes; (void)n_in; (void)out_size; (void)ws_size;

  const float* x    = (const float*)d_in[0];
  const float* w_f1 = (const float*)d_in[1];
  const float* b_f1 = (const float*)d_in[2];
  const float* w_f2 = (const float*)d_in[3];
  const float* b_f2 = (const float*)d_in[4];
  const float* w_v1 = (const float*)d_in[5];
  const float* b_v1 = (const float*)d_in[6];
  const float* w_v2 = (const float*)d_in[7];
  const float* b_v2 = (const float*)d_in[8];
  const float* w_q  = (const float*)d_in[9];
  const float* b_q  = (const float*)d_in[10];
  const float* w_k  = (const float*)d_in[11];
  const float* b_k  = (const float*)d_in[12];
  float* out = (float*)d_out;

  // Workspace carve-up (all offsets 256B aligned).
  char* ws = (char*)d_ws;
  size_t off = 0;
  auto carve = [&](size_t bytes) {
    void* p = ws + off;
    off += (bytes + 255) & ~(size_t)255;
    return p;
  };
  float*     q_buf  = (float*)carve((size_t)BB * QC * LL * 4);    // 2 MB
  float*     k_buf  = (float*)carve((size_t)BB * QC * LL * 4);    // 2 MB
  float*     v1_buf = (float*)carve((size_t)BB * CC * LL * 4);    // 4 MB
  float*     v_buf  = (float*)carve((size_t)BB * CC * LL * 4);    // 4 MB
  float*     f1_buf = (float*)carve((size_t)BB * CC * LL * 4);    // 4 MB
  float*     f2_buf = (float*)carve((size_t)BB * CC * LL * 4);    // 4 MB
  _Float16*  Qmat   = (_Float16*)carve((size_t)BB * LL * DD * 2); // 9.4 MB
  _Float16*  Kmat   = (_Float16*)carve((size_t)BB * LL * DD * 2); // 9.4 MB
  float*     pVal   = (float*)carve((size_t)BB * NSEG * LL * 4);  // 256 KB
  int*       pIdx   = (int*)carve((size_t)BB * NSEG * LL * 4);    // 256 KB
  float*     Rmax   = (float*)carve((size_t)BB * LL * 4);
  int*       Ridx   = (int*)carve((size_t)BB * LL * 4);

  const int TPB = 256;
  const int gridQK = (BB * QC * LL) / TPB;   // 2048
  const int gridC  = (BB * CC * LL) / TPB;   // 4096

  // Convolutions (q, k, v-path, f-path)
  conv3_kernel<<<gridQK, TPB, 0, stream>>>(x, nullptr, w_q,  b_q,  q_buf,  CC, QC, 0);
  conv3_kernel<<<gridQK, TPB, 0, stream>>>(x, nullptr, w_k,  b_k,  k_buf,  CC, QC, 0);
  conv3_kernel<<<gridC,  TPB, 0, stream>>>(x, nullptr, w_v1, b_v1, v1_buf, CC, CC, 0);
  conv3_kernel<<<gridC,  TPB, 0, stream>>>(v1_buf, x,  w_v2, b_v2, v_buf,  CC, CC, 0);
  conv3_kernel<<<gridC,  TPB, 0, stream>>>(x, nullptr, w_f1, b_f1, f1_buf, CC, CC, 1);
  conv3_kernel<<<gridC,  TPB, 0, stream>>>(f1_buf, nullptr, w_f2, b_f2, f2_buf, CC, CC, 1);

  // unfold + normalize + pack to f16 row-major [B][L][288]
  unfold_norm_f16_kernel<<<(BB * LL) / TPB, TPB, 0, stream>>>(q_buf, Qmat);
  unfold_norm_f16_kernel<<<(BB * LL) / TPB, TPB, 0, stream>>>(k_buf, Kmat);

  // Fused WMMA affinity + segment max/argmax: 256 blocks x 8 waves
  attn_argmax_kernel<<<BB * (LL / 256) * NSEG, 256, 0, stream>>>(Qmat, Kmat, pVal, pIdx);
  reduce_seg_kernel<<<(BB * LL) / TPB, TPB, 0, stream>>>(pVal, pIdx, Rmax, Ridx);

  // Gather/fold epilogue + residual combine
  epilogue_kernel<<<gridC, TPB, 0, stream>>>(x, v_buf, f2_buf, Rmax, Ridx, out);
}
